// GraphGDP_83167746720457
// MI455X (gfx1250) — matrix operand order repaired and checked by reading, not compiled, not run on hardware
//
#include <hip/hip_runtime.h>
#include <cstdint>

// ---------------- problem constants (from the reference) ----------------
#define NN   10000
#define EE1  320000
#define EE2  160000
#define HH   128
#define LL   3
#define BB   16

// GEMM flags
#define GF_RELU 1   // relu epilogue
#define GF_A16  2   // A operand stored as f16
#define GF_O16  4   // store output as f16

typedef __attribute__((ext_vector_type(16))) _Float16 v16h;
typedef __attribute__((ext_vector_type(8)))  _Float16 v8h;
typedef __attribute__((ext_vector_type(8)))  float    v8f;
typedef __attribute__((ext_vector_type(4)))  unsigned u32x4;
typedef __attribute__((ext_vector_type(8)))  int      i32x8;
typedef __attribute__((ext_vector_type(4)))  int      i32x4;

// Scatter one f32 weight element into the LDS f16 B-fragment layout for
// v_wmma_f32_16x16x32_f16: fragment = (chunk,kstep,ntile); within a fragment
// lane = col | ((k>>4)&1)<<4 and the 16 halves per lane are K-consecutive,
// so the GEMM inner loop reads one contiguous 32B LDS chunk per fragment.
__device__ __forceinline__ void store_wfrag(_Float16* sW, int k, int n, float w)
{
  const int chunk = k >> 7, kc = k & 127;
  const int kstep = kc >> 5, kk = kc & 31;
  const int ntile = n >> 4,  c  = n & 15;
  const int lane = c | (((kk >> 4) & 1) << 4);
  const int h    = kk & 15;
  const int frag = (chunk << 5) | (kstep << 3) | ntile;   // 32 frags / 128-chunk
  sW[(((frag << 5) | lane) << 4) | h] = (_Float16)w;
}

// =======================================================================
// WMMA GEMM: out[M,128] = act( [A0|A1][M,K] @ W[K,128] + bias + residual )
//   K = 128 (A0 only) or 256 (A0|A1 = the two concat halves).
// Weights: staged into LDS via the Tensor Data Mover when the toolchain
// exposes __builtin_amdgcn_tensor_load_to_lds (wave 0 issues the DMA,
// s_wait_tensorcnt 0, barrier, then cooperative f32->f16 swizzle);
// otherwise a direct cooperative global load path is used.
// Each wave owns a 16x128 output stripe: 8 f32 accumulators (v8f each).
// =======================================================================
__global__ __launch_bounds__(256) void wmma_gemm_kernel(
    const void* __restrict__ A0v, const void* __restrict__ A1v,
    const float* __restrict__ W,  const float* __restrict__ bias,
    const float* __restrict__ residual, void* __restrict__ outv,
    int M, int K, int flags)
{
  extern __shared__ __align__(16) char smem_raw[];
  _Float16* sW = reinterpret_cast<_Float16*>(smem_raw);

  const int tid = threadIdx.x;
  const int nchunks = K >> 7;

#if __has_builtin(__builtin_amdgcn_tensor_load_to_lds)
  // ---- TDM weight staging: DMA one 128x128 f32 chunk into LDS raw region ----
  {
    float* sRaw = reinterpret_cast<float*>(smem_raw + (size_t)nchunks * 32768);
    for (int chunk = 0; chunk < nchunks; ++chunk) {
      if (tid < 32) {                       // wave 0 issues the tensor DMA
        const uint64_t ga  = (uint64_t)(uintptr_t)(W + ((size_t)chunk << 14));
        const unsigned lds = (unsigned)(uintptr_t)sRaw;     // LDS byte offset
        const unsigned d0  = 16384u;        // 128*128 elements, one row
        u32x4 g0; i32x8 g1;
        i32x4 gz4 = (i32x4){0, 0, 0, 0};
        i32x8 gz8 = (i32x8){0, 0, 0, 0, 0, 0, 0, 0};
        g0[0] = 1u;                         // count=1 (valid user descriptor)
        g0[1] = lds;                        // lds_addr
        g0[2] = (unsigned)ga;               // global_addr[31:0]
        g0[3] = (unsigned)((ga >> 32) & 0x01FFFFFFu) | (2u << 30); // addr[56:32] | type=2
        g1[0] = (int)(2u << 16);            // data_size = 4 bytes
        g1[1] = (int)((d0 & 0xFFFFu) << 16);            // tensor_dim0[15:0]
        g1[2] = (int)(((d0 >> 16) & 0xFFFFu) | (1u << 16)); // tensor_dim0[31:16] | tensor_dim1=1
        g1[3] = (int)(d0 << 16);            // tensor_dim1 hi=0 | tile_dim0 = 16384
        g1[4] = 1;                          // tile_dim1 = 1, tile_dim2 = 0
        g1[5] = (int)d0;                    // tensor_dim0_stride[31:0]
        g1[6] = 0;                          // stride0 hi | stride1 lo
        g1[7] = 0;
        __builtin_amdgcn_tensor_load_to_lds(g0, g1, gz4, gz4, gz8, 0);
        __builtin_amdgcn_s_wait_tensorcnt(0);
      }
      __syncthreads();                      // raw chunk visible to all waves
      for (int idx = tid; idx < 16384; idx += 256)
        store_wfrag(sW, (chunk << 7) | (idx >> 7), idx & 127, sRaw[idx]);
      __syncthreads();                      // before raw region is re-used
    }
  }
#else
  // ---- fallback: direct cooperative global load + swizzle ----
  {
    const int nElem = K << 7;
    for (int idx = tid; idx < nElem; idx += 256)
      store_wfrag(sW, idx >> 7, idx & 127, W[idx]);
    __syncthreads();
  }
#endif

  const int wave = tid >> 5, lane = tid & 31;
  const int r  = lane & 15;                 // A row within tile / D column
  const int kh = lane >> 4;                 // A K-half selector
  const int mtiles = M >> 4;
  const int a16 = flags & GF_A16;
  const int esz = a16 ? 2 : 4;

  for (int rt = blockIdx.x * 8 + wave; rt < mtiles; rt += gridDim.x * 8) {
    const int row = (rt << 4) + r;
    v8f acc[8];
#pragma unroll
    for (int t = 0; t < 8; ++t) acc[t] = (v8f){0.f,0.f,0.f,0.f,0.f,0.f,0.f,0.f};

    // prefetch next stripe of A (global_prefetch_b8)
    {
      const int rtn = rt + gridDim.x * 8;
      if (rtn < mtiles) {
        const size_t po = (size_t)((rtn << 4) + r) * HH * esz;
        __builtin_prefetch((const char*)A0v + po, 0, 0);
        if (nchunks > 1) __builtin_prefetch((const char*)A1v + po, 0, 0);
      }
    }

    for (int chunk = 0; chunk < nchunks; ++chunk) {
      const void* __restrict__ A = chunk ? A1v : A0v;
#pragma unroll
      for (int ks = 0; ks < 4; ++ks) {
        // A 16x32 f16 frag: lane row = lane&15; halves 0..15 = K kb..kb+15,
        // kb = ks*32 + (lane>>4)*16.
        const int kb = (ks << 5) + (kh << 4);
        union { v16h v; v8h g[2]; _Float16 h[16]; } af;
        if (a16) {
          const _Float16* arow = (const _Float16*)A + (size_t)row * HH;
          af.g[0] = *(const v8h*)(arow + kb);
          af.g[1] = *(const v8h*)(arow + kb + 8);
        } else {
          const float* arow = (const float*)A + (size_t)row * HH;
          const float4 x0 = *(const float4*)(arow + kb);
          const float4 x1 = *(const float4*)(arow + kb + 4);
          const float4 x2 = *(const float4*)(arow + kb + 8);
          const float4 x3 = *(const float4*)(arow + kb + 12);
          af.h[0]=(_Float16)x0.x; af.h[1]=(_Float16)x0.y; af.h[2]=(_Float16)x0.z; af.h[3]=(_Float16)x0.w;
          af.h[4]=(_Float16)x1.x; af.h[5]=(_Float16)x1.y; af.h[6]=(_Float16)x1.z; af.h[7]=(_Float16)x1.w;
          af.h[8]=(_Float16)x2.x; af.h[9]=(_Float16)x2.y; af.h[10]=(_Float16)x2.z; af.h[11]=(_Float16)x2.w;
          af.h[12]=(_Float16)x3.x; af.h[13]=(_Float16)x3.y; af.h[14]=(_Float16)x3.z; af.h[15]=(_Float16)x3.w;
        }
        const _Float16* sB = sW + ((size_t)((((chunk << 5) | (ks << 3)) << 5) | lane) << 4);
#pragma unroll
        for (int nt = 0; nt < 8; ++nt) {
          const v16h bf = *(const v16h*)(sB + ((size_t)nt << 9));  // frag stride 512 halves
          acc[nt] = __builtin_amdgcn_wmma_f32_16x16x32_f16(
              false, af.v, false, bf, (short)0, acc[nt], false, false);
        }
      }
    }

    // Epilogue. D layout: col = lane&15 (+16*ntile); VGPR j -> row j + 8*(lane>>4)
    const int rOff = kh << 3;
#pragma unroll
    for (int nt = 0; nt < 8; ++nt) {
      const int col = (nt << 4) + r;
      const float bcol = bias ? bias[col] : 0.f;
#pragma unroll
      for (int j = 0; j < 8; ++j) {
        const int orow = (rt << 4) + rOff + j;
        const size_t o = (size_t)orow * HH + col;
        float v = acc[nt][j] + bcol;
        if (residual) v += residual[o];
        if (flags & GF_RELU) v = fmaxf(v, 0.f);
        if (flags & GF_O16) ((_Float16*)outv)[o] = (_Float16)v;
        else                ((float*)outv)[o]    = v;
      }
    }
  }
}

// ---------------- small helper kernels ----------------

// out[m, :] = act(x[m*stride] * w[:] + b[:])   (rank-1 first MLP layer, din==1)
__global__ void rank1_kernel(const float* __restrict__ x, int xstride,
                             const float* __restrict__ w, const float* __restrict__ b,
                             void* __restrict__ out, int M, int relu, int of16)
{
  const int idx = blockIdx.x * 256 + threadIdx.x;
  const int m = idx >> 7, c = idx & 127;
  if (m >= M) return;
  float v = fmaf(x[(size_t)m * xstride], w[c], b[c]);
  if (relu) v = fmaxf(v, 0.f);
  if (of16) ((_Float16*)out)[idx] = (_Float16)v;
  else      ((float*)out)[idx]    = v;
}

__global__ void tnodes_kernel(const float* __restrict__ t, const int* __restrict__ batch,
                              float* __restrict__ tn, int n)
{
  const int i = blockIdx.x * 256 + threadIdx.x;
  if (i < n) tn[i] = t[batch[i]];
}

__global__ void degemb_kernel(const float* __restrict__ node_x, const float* __restrict__ table,
                              float* __restrict__ de, int n)
{
  const int idx = blockIdx.x * 256 + threadIdx.x;
  const int m = idx >> 7, c = idx & 127;
  if (m >= n) return;
  const int di = (int)node_x[(size_t)m * 2 + 1];
  de[idx] = table[(size_t)di * HH + c];
}

__global__ void cnt_kernel(const int* __restrict__ batch, float* __restrict__ cnt, int n)
{
  const int i = blockIdx.x * 256 + threadIdx.x;
  if (i < n) atomicAdd(&cnt[batch[i]], 1.0f);
}

// ---- segment softmax over dst, ordered-uint key for float atomicMax ----
__device__ __forceinline__ unsigned fkey(float f) {
  unsigned u = __float_as_uint(f);
  return (u & 0x80000000u) ? ~u : (u | 0x80000000u);
}
__device__ __forceinline__ float funkey(unsigned k) {
  unsigned u = (k & 0x80000000u) ? (k & 0x7fffffffu) : ~k;
  return __uint_as_float(u);
}

// wave-per-edge: logit = dot(q[dst], k[src]+e[edge]) / sqrt(H); seg max by dst
__global__ __launch_bounds__(256) void attn_logits_kernel(
    const float* __restrict__ q, const float* __restrict__ k, const float* __restrict__ e,
    const int* __restrict__ src, const int* __restrict__ dst,
    float* __restrict__ elog, unsigned* __restrict__ segmax, int E)
{
  const int wid  = (blockIdx.x * 256 + threadIdx.x) >> 5;
  const int lane = threadIdx.x & 31;
  if (wid >= E) return;
  const int s = src[wid], d = dst[wid];
  const float* qp = q + (size_t)d * HH + lane * 4;
  const float* kp = k + (size_t)s * HH + lane * 4;
  const float* ep = e + (size_t)wid * HH + lane * 4;
  float sum = 0.f;
#pragma unroll
  for (int i = 0; i < 4; ++i) sum += qp[i] * (kp[i] + ep[i]);
#pragma unroll
  for (int off = 16; off; off >>= 1) sum += __shfl_xor(sum, off, 32);
  if (lane == 0) {
    const float lg = sum * 0.08838834764831845f;   // 1/sqrt(128)
    elog[wid] = lg;
    atomicMax(&segmax[d], fkey(lg));
  }
}

__global__ void attn_exp_kernel(const int* __restrict__ dst, const unsigned* __restrict__ segmax,
                                float* __restrict__ elog, float* __restrict__ segsum, int E)
{
  const int e = blockIdx.x * 256 + threadIdx.x;
  if (e >= E) return;
  const int d = dst[e];
  const float ex = __expf(elog[e] - funkey(segmax[d]));
  elog[e] = ex;
  atomicAdd(&segsum[d], ex);
}

// msg = alpha * (v[src] + e); out[dst] += msg   (thread = edge x 4 channels)
__global__ void msg_agg_kernel(const int* __restrict__ src, const int* __restrict__ dst,
                               const float* __restrict__ elog, const float* __restrict__ segsum,
                               const float* __restrict__ v, const float* __restrict__ e,
                               float* __restrict__ msg, float* __restrict__ out, int E)
{
  const int idx = blockIdx.x * 256 + threadIdx.x;
  const int ed = idx >> 5;
  if (ed >= E) return;
  const int g = (idx & 31) << 2;
  const int s = src[ed], d = dst[ed];
  const float a = elog[ed] / (segsum[d] + 1e-16f);
  const size_t eo = (size_t)ed * HH + g, so = (size_t)s * HH + g, don = (size_t)d * HH + g;
#pragma unroll
  for (int i = 0; i < 4; ++i) {
    const float m = a * (v[so + i] + e[eo + i]);
    msg[eo + i] = m;
    atomicAdd(&out[don + i], m);
  }
}

// ---- GraphNorm (3 passes; meansum/varsum pre-zeroed, cnt precomputed) ----
__global__ void gn_mean_kernel(const float* __restrict__ x, const int* __restrict__ batch,
                               float* __restrict__ meansum, int n)
{
  const int idx = blockIdx.x * 256 + threadIdx.x;
  const int m = idx >> 7, c = idx & 127;
  if (m >= n) return;
  atomicAdd(&meansum[batch[m] * HH + c], x[idx]);
}

__global__ void gn_subvar_kernel(float* __restrict__ x, const int* __restrict__ batch,
                                 const float* __restrict__ meansum, const float* __restrict__ cnt,
                                 const float* __restrict__ ms, float* __restrict__ varsum, int n)
{
  const int idx = blockIdx.x * 256 + threadIdx.x;
  const int m = idx >> 7, c = idx & 127;
  if (m >= n) return;
  const int b = batch[m];
  const float mean = meansum[b * HH + c] / cnt[b];
  const float sub = x[idx] - ms[c] * mean;
  x[idx] = sub;
  atomicAdd(&varsum[b * HH + c], sub * sub);
}

__global__ void gn_apply_kernel(float* __restrict__ x, const int* __restrict__ batch,
                                const float* __restrict__ varsum, const float* __restrict__ cnt,
                                const float* __restrict__ w, const float* __restrict__ bb, int n)
{
  const int idx = blockIdx.x * 256 + threadIdx.x;
  const int m = idx >> 7, c = idx & 127;
  if (m >= n) return;
  const int b = batch[m];
  const float var = varsum[b * HH + c] / cnt[b];
  x[idx] = fmaf(w[c], x[idx] * rsqrtf(var + 1e-5f), bb[c]);
}

// decoder final layer (128 -> 1): wave-per-edge dot product (f16 or f32 input)
__global__ __launch_bounds__(256) void dot_out_kernel(
    const void* __restrict__ h, const float* __restrict__ w, const float* __restrict__ b,
    float* __restrict__ out, int E, int hf16)
{
  const int wid  = (blockIdx.x * 256 + threadIdx.x) >> 5;
  const int lane = threadIdx.x & 31;
  if (wid >= E) return;
  const float* wp = w + lane * 4;
  float sum = 0.f;
  if (hf16) {
    const _Float16* hp = (const _Float16*)h + (size_t)wid * HH + lane * 4;
#pragma unroll
    for (int i = 0; i < 4; ++i) sum += (float)hp[i] * wp[i];
  } else {
    const float* hp = (const float*)h + (size_t)wid * HH + lane * 4;
#pragma unroll
    for (int i = 0; i < 4; ++i) sum += hp[i] * wp[i];
  }
#pragma unroll
  for (int off = 16; off; off >>= 1) sum += __shfl_xor(sum, off, 32);
  if (lane == 0) out[wid] = sum + b[0];
}

// =======================================================================
// Host orchestration
// =======================================================================
extern "C" void kernel_launch(void* const* d_in, const int* in_sizes, int n_in,
                              void* d_out, int out_size, void* d_ws, size_t ws_size,
                              hipStream_t stream)
{
  (void)in_sizes; (void)n_in; (void)out_size; (void)ws_size;

  // ---- input indexing (params pytree flattened in dict-literal order:
  // MLP = [W1,b1,W2,b2,W3,b3]; tconv = q(W,b),k,v,e,skip; gn = (w,b,ms)) ----
  auto F = [&](int i) -> const float* { return (const float*)d_in[i]; };
  const float* node_x = F(0);
  const int*   ei1    = (const int*)d_in[1];
  const float* ea1    = F(2);
  const int*   ei2    = (const int*)d_in[3];
  const float* ea2    = F(4);
  const int*   batch  = (const int*)d_in[5];
  const float* tval   = F(6);
  int p = 7;
  const int enc_part = p; p += 6;
  const int enc_full = p; p += 6;
  const int time_mlp = p; p += 6;
  const int node_mlp = p; p += 6;   // W1 is [129,128]
  const int dec_mlp  = p; p += 6;   // W1 [256,128], W2 [128,128], W3 [128,1]
  const int deg_tab  = p; p += 1;
  const int gnn_g    = p; p += 30;  // 3 layers x (qW,qb,kW,kb,vW,vb,eW,eb,sW,sb)
  const int gnn_f    = p; p += 30;
  const int mlp_int  = p; p += 18;
  const int gnf      = p; p += 9;   // 3 x (w,b,ms)
  const int gnp      = p; p += 9;

  const int* src1 = ei1;  const int* dst1 = ei1 + EE1;
  const int* src2 = ei2;  const int* dst2 = ei2 + EE2;

  // ---- workspace bump allocator ----
  char* wsb = (char*)d_ws;
  size_t off = 0;
  auto alloc = [&](size_t nfloats) -> float* {
    float* r = (float*)(wsb + off);
    off = (off + nfloats * sizeof(float) + 255) & ~(size_t)255;
    return r;
  };
  const size_t NH = (size_t)NN * HH, E1H = (size_t)EE1 * HH, E2H = (size_t)EE2 * HH;
  float* t_nodes = alloc(NN);
  float* cnt     = alloc(BB);
  float* meansum = alloc(BB * HH);
  float* varsum  = alloc(BB * HH);
  unsigned* segmax = (unsigned*)alloc(NN);
  float* segsum  = alloc(NN);
  float* elog1   = alloc(EE1);
  float* elog2   = alloc(EE2);
  float* deg_emb = alloc(NH);
  float* tmpA = alloc(NH);  float* tmpB = alloc(NH);
  float* nf0a = alloc(NH);  float* nf0b = alloc(NH);
  float* nf1a = alloc(NH);  float* nf1b = alloc(NH);
  float* qb = alloc(NH); float* kb = alloc(NH); float* vb = alloc(NH);
  float* I1 = alloc(E1H); float* X1 = alloc(E1H); float* Z1 = alloc(E1H);
  float* I2 = alloc(E2H); float* P2 = alloc(E2H); float* Q2 = alloc(E2H); float* Z2 = alloc(E2H);

  auto gemm = [&](const void* A0, const void* A1, const float* W, const float* bias,
                  const float* resid, void* outp, int M, int K, int flags) {
    const int mt = M >> 4;
    const int blocks = (mt + 7) / 8;
    // f16 fragment region (+32KB per 128-chunk) + 64KB TDM raw staging region
    const size_t sh = (size_t)(K >> 7) * 32768 + 65536;
    wmma_gemm_kernel<<<dim3(blocks), dim3(256), sh, stream>>>(A0, A1, W, bias, resid, outp, M, K, flags);
  };
  auto ew = [&](int total) { return dim3((total + 255) / 256); };

  // ---------------- stage A: node features ----------------
  tnodes_kernel<<<ew(NN), 256, 0, stream>>>(tval, batch, t_nodes, NN);
  degemb_kernel<<<ew(NN * HH), 256, 0, stream>>>(node_x, F(deg_tab), deg_emb, NN);
  (void)hipMemsetAsync(cnt, 0, BB * sizeof(float), stream);
  cnt_kernel<<<ew(NN), 256, 0, stream>>>(batch, cnt, NN);

  // time encoder -> nf0a (hidden activations stored f16)
  rank1_kernel<<<ew(NN * HH), 256, 0, stream>>>(t_nodes, 1, F(time_mlp), F(time_mlp + 1), tmpA, NN, 1, 1);
  gemm(tmpA, nullptr, F(time_mlp + 2), F(time_mlp + 3), nullptr, tmpB, NN, 128, GF_RELU | GF_A16 | GF_O16);
  gemm(tmpB, nullptr, F(time_mlp + 4), F(time_mlp + 5), nullptr, nf0a, NN, 128, GF_A16);

  // node encoder ([feat, deg_emb] @ W[129,128]) -> nf0b
  rank1_kernel<<<ew(NN * HH), 256, 0, stream>>>(node_x, 2, F(node_mlp), F(node_mlp + 1), tmpA, NN, 0, 0);
  gemm(deg_emb, nullptr, F(node_mlp) + HH, nullptr, tmpA, tmpB, NN, 128, GF_RELU | GF_O16); // rows 1..128
  gemm(tmpB, nullptr, F(node_mlp + 2), F(node_mlp + 3), nullptr, tmpA, NN, 128, GF_RELU | GF_A16 | GF_O16);
  gemm(tmpA, nullptr, F(node_mlp + 4), F(node_mlp + 5), nullptr, nf0b, NN, 128, GF_A16);

  // ---------------- stage B: edge encoders (f16 hiddens) ----------------
  rank1_kernel<<<ew(EE1 * HH), 256, 0, stream>>>(ea1, 1, F(enc_full), F(enc_full + 1), Z1, EE1, 1, 1);
  gemm(Z1, nullptr, F(enc_full + 2), F(enc_full + 3), nullptr, X1, EE1, 128, GF_RELU | GF_A16 | GF_O16);
  gemm(X1, nullptr, F(enc_full + 4), F(enc_full + 5), nullptr, I1, EE1, 128, GF_A16);
  rank1_kernel<<<ew(EE2 * HH), 256, 0, stream>>>(ea2, 1, F(enc_part), F(enc_part + 1), Z2, EE2, 1, 1);
  gemm(Z2, nullptr, F(enc_part + 2), F(enc_part + 3), nullptr, P2, EE2, 128, GF_RELU | GF_A16 | GF_O16);
  gemm(P2, nullptr, F(enc_part + 4), F(enc_part + 5), nullptr, I2, EE2, 128, GF_A16);

  // ---------------- stage C: L layers of dual tconv ----------------
  float* nfa = nf0a; float* nfb = nf0b; float* ofa = nf1a; float* ofb = nf1b;
  const float* ee1cur = I1;
  const float* ee2cur = I2;

  auto tconv_graph = [&](int pb, const float* ee, float* eedge, float* msgbuf, float* elog,
                         const int* src, const int* dst, int E, float* outp) {
    gemm(nfa, nfb, F(pb + 0), F(pb + 1), nullptr, qb,   NN, 256, 0);   // q
    gemm(nfa, nfb, F(pb + 2), F(pb + 3), nullptr, kb,   NN, 256, 0);   // k
    gemm(nfa, nfb, F(pb + 4), F(pb + 5), nullptr, vb,   NN, 256, 0);   // v
    gemm(nfa, nfb, F(pb + 8), F(pb + 9), nullptr, outp, NN, 256, 0);   // skip -> out (agg added)
    gemm(ee, nullptr, F(pb + 6), F(pb + 7), nullptr, eedge, E, 128, 0); // e = ee@We + be
    (void)hipMemsetAsync(segmax, 0, NN * sizeof(unsigned), stream);
    (void)hipMemsetAsync(segsum, 0, NN * sizeof(float), stream);
    attn_logits_kernel<<<ew(E * 32), 256, 0, stream>>>(qb, kb, eedge, src, dst, elog, segmax, E);
    attn_exp_kernel<<<ew(E), 256, 0, stream>>>(dst, segmax, elog, segsum, E);
    msg_agg_kernel<<<ew(E * 32), 256, 0, stream>>>(src, dst, elog, segsum, vb, eedge, msgbuf, outp, E);
  };
  auto gnorm = [&](float* x, int gbase) {
    (void)hipMemsetAsync(meansum, 0, BB * HH * sizeof(float), stream);
    (void)hipMemsetAsync(varsum, 0, BB * HH * sizeof(float), stream);
    gn_mean_kernel<<<ew(NN * HH), 256, 0, stream>>>(x, batch, meansum, NN);
    gn_subvar_kernel<<<ew(NN * HH), 256, 0, stream>>>(x, batch, meansum, cnt, F(gbase + 2), varsum, NN);
    gn_apply_kernel<<<ew(NN * HH), 256, 0, stream>>>(x, batch, varsum, cnt, F(gbase), F(gbase + 1), NN);
  };

  for (int i = 0; i < LL; ++i) {
    // graph 1 (full): msg1 -> X1 (becomes ee1); out1 -> ofa
    tconv_graph(gnn_g + i * 10, ee1cur, Z1, X1, elog1, src1, dst1, EE1, ofa);
    gnorm(ofa, gnf + i * 3);
    // graph 2 (partial): msg2 -> P2; out2 -> ofb
    tconv_graph(gnn_f + i * 10, ee2cur, Z2, P2, elog2, src2, dst2, EE2, ofb);
    gnorm(ofb, gnp + i * 3);
    // ee1 = msg1
    ee1cur = X1;
    // ee2 = mlp_interactions(msg2) + ee2_init   (f16 hiddens)
    const int mb = mlp_int + i * 6;
    gemm(P2, nullptr, F(mb + 0), F(mb + 1), nullptr, Z2, EE2, 128, GF_RELU | GF_O16);           // t1
    gemm(Z2, nullptr, F(mb + 2), F(mb + 3), nullptr, Q2, EE2, 128, GF_RELU | GF_A16 | GF_O16);  // t2
    gemm(Q2, nullptr, F(mb + 4), F(mb + 5), I2, P2, EE2, 128, GF_A16);                          // + ee2_init
    ee2cur = P2;
    // nf = concat(out1, out2): swap ping-pong node buffers
    float* ta = nfa; nfa = ofa; ofa = ta;
    float* tb = nfb; nfb = ofb; ofb = tb;
  }

  // ---------------- stage D: decoder on concat(ee1, ee1_init) ----------------
  gemm(ee1cur, I1, F(dec_mlp + 0), F(dec_mlp + 1), nullptr, Z1, EE1, 256, GF_RELU | GF_O16);
  gemm(Z1, nullptr, F(dec_mlp + 2), F(dec_mlp + 3), nullptr, X1, EE1, 128, GF_RELU | GF_A16 | GF_O16);
  dot_out_kernel<<<ew(EE1 * 32), 256, 0, stream>>>(X1, F(dec_mlp + 4), F(dec_mlp + 5),
                                                   (float*)d_out, EE1, 1);
}
